// ODST_1675037245855
// MI455X (gfx1250) — matrix operand order repaired and checked by reading, not compiled, not run on hardware
//
#include <hip/hip_runtime.h>
#include <hip/hip_bf16.h>

typedef __attribute__((ext_vector_type(8)))  float  v8f;
typedef __attribute__((ext_vector_type(16))) __bf16 v16bf;
typedef __attribute__((ext_vector_type(8)))  __bf16 v8bf;

#define DEPTH_      6
#define FDIM        256     // IN_FEATURES == K
#define NTREES      256
#define NTOT        1536    // NTREES * DEPTH_
#define BATCH_      2048
#define LEAVES      64      // 2^DEPTH_

#define TILE_M      16      // batch rows per block
#define TPB_TREES   16      // trees per block
#define TILE_N      96      // TPB_TREES * DEPTH_

// ---------------------------------------------------------------------------
// Kernel 1: sparsemax over the feature axis for one (tree,depth) column.
// 1536 blocks x 256 threads. Bitonic sort (descending) + inclusive scan in LDS.
// ---------------------------------------------------------------------------
__global__ __launch_bounds__(256) void sparsemax_kernel(
    const float* __restrict__ logits,   // [F, NTOT]
    float* __restrict__ fs)             // [F, NTOT]
{
    __shared__ float sv[256];   // values -> sorted values
    __shared__ float scs[256];  // inclusive cumsum of sorted values
    __shared__ float red[256];  // reduction scratch

    const int n = blockIdx.x;     // column in [0, 1536)
    const int f = threadIdx.x;    // feature row in [0, 256)

    const float v = logits[f * NTOT + n];

    // max-reduce
    red[f] = v;
    __syncthreads();
    #pragma unroll
    for (int s = 128; s > 0; s >>= 1) {
        if (f < s) red[f] = fmaxf(red[f], red[f + s]);
        __syncthreads();
    }
    const float mx = red[0];
    __syncthreads();

    const float xsv = v - mx;
    sv[f] = xsv;
    __syncthreads();

    // bitonic sort, descending
    for (int k = 2; k <= 256; k <<= 1) {
        for (int j = k >> 1; j > 0; j >>= 1) {
            const int ixj = f ^ j;
            if (ixj > f) {
                const float a = sv[f];
                const float b = sv[ixj];
                const bool desc = ((f & k) == 0);
                if (desc ? (a < b) : (a > b)) { sv[f] = b; sv[ixj] = a; }
            }
            __syncthreads();
        }
    }

    // inclusive scan (Hillis-Steele)
    const float z = sv[f];
    scs[f] = z;
    __syncthreads();
    for (int off = 1; off < 256; off <<= 1) {
        const float add = (f >= off) ? scs[f - off] : 0.0f;
        __syncthreads();
        scs[f] += add;
        __syncthreads();
    }

    const float cs = scs[f] - 1.0f;               // topk_cumsum - 1
    // support count: rho * z > cs  (monotone prefix property)
    red[f] = (((float)(f + 1)) * z > cs) ? 1.0f : 0.0f;
    __syncthreads();
    #pragma unroll
    for (int s = 128; s > 0; s >>= 1) {
        if (f < s) red[f] += red[f + s];
        __syncthreads();
    }
    const int ksup = (int)red[0];
    const float tau = (scs[ksup - 1] - 1.0f) / (float)ksup;

    fs[f * NTOT + n] = fmaxf(xsv - tau, 0.0f);
}

// ---------------------------------------------------------------------------
// Kernel 2: fused  fv = x @ fs  (bf16 WMMA)  ->  gates -> leaf products -> out
// grid (BATCH/16, NTOT/96), 256 threads (8 waves; waves 0..5 run WMMA).
// Dynamic LDS (57344 B):
//   phase 1: xs  bf16[16][256]  (8 KB)   | fst bf16[96][256] (48 KB)
//   phase 2 (aliased): fvt f32[16][96] | resp f32[16][64] | thr[96] | itmp[96]
// ---------------------------------------------------------------------------
__global__ __launch_bounds__(256) void odst_fused_kernel(
    const float* __restrict__ x,          // [B, F]
    const float* __restrict__ fs,         // [F, NTOT]
    const float* __restrict__ response,   // [T, 1, 64]
    const float* __restrict__ thresholds, // [T, 6]
    const float* __restrict__ log_temps,  // [T, 6]
    float* __restrict__ out)              // [B, T]
{
    extern __shared__ char smem[];
    __bf16* xs  = (__bf16*)smem;          // [TILE_M][FDIM]
    __bf16* fst = xs + TILE_M * FDIM;     // [TILE_N][FDIM]   (transposed fs tile)

    const int tid    = threadIdx.x;
    const int b_base = blockIdx.x * TILE_M;
    const int t_base = blockIdx.y * TPB_TREES;
    const int n_base = t_base * DEPTH_;

    // ---- phase 1: stage A (x rows) and B (fs cols, transposed) as bf16 ----
    for (int i = tid; i < TILE_M * FDIM; i += 256) {
        const int m = i >> 8, k = i & 255;
        xs[i] = (__bf16)x[(b_base + m) * FDIM + k];
    }
    for (int i = tid; i < TILE_N * FDIM; i += 256) {
        const int k = i / TILE_N, nl = i % TILE_N;     // contiguous global reads
        fst[nl * FDIM + k] = (__bf16)fs[k * NTOT + n_base + nl];
    }
    __syncthreads();

    // ---- WMMA: waves 0..5 each own a 16x16 fv subtile, K = 256 ----
    v8f acc = {};
    const int wave = tid >> 5;
    const int lane = tid & 31;
    if (wave < 6) {
        const int m  = lane & 15;       // A row / B column within subtile
        const int hi = lane >> 4;       // lane-group select
        const __bf16* arow = xs  + m * FDIM + hi * 8;                 // A: 16x32 bf16 layout
        const __bf16* brow = fst + (wave * 16 + m) * FDIM + hi * 8;   // B: 32x16 bf16 layout
        #pragma unroll
        for (int kb = 0; kb < FDIM; kb += 32) {
            union { v16bf v; v8bf h[2]; } A, B;
            A.h[0] = *(const v8bf*)(arow + kb);        // K = base+0..7
            A.h[1] = *(const v8bf*)(arow + kb + 16);   // K = base+16..23
            B.h[0] = *(const v8bf*)(brow + kb);        // K = base+0..7
            B.h[1] = *(const v8bf*)(brow + kb + 16);   // K = base+16..23
            acc = __builtin_amdgcn_wmma_f32_16x16x32_bf16(
                      false, A.v, false, B.v, (short)0, acc, false, false);
        }
    }
    __syncthreads();

    // ---- phase 2: alias LDS ----
    float* fvt  = (float*)smem;                 // [TILE_M][TILE_N]
    float* resp = fvt  + TILE_M * TILE_N;       // [TPB_TREES][64]
    float* thrS = resp + TPB_TREES * LEAVES;    // [TILE_N]
    float* itmp = thrS + TILE_N;                // [TILE_N]

    if (wave < 6) {
        const int nl = wave * 16 + (lane & 15);
        const int m0 = (lane >> 4) * 8;         // D layout: VGPR r -> M = r + 8*hi
        #pragma unroll
        for (int r = 0; r < 8; ++r)
            fvt[(m0 + r) * TILE_N + nl] = acc[r];
    }
    for (int i = tid; i < TPB_TREES * LEAVES; i += 256)
        resp[i] = response[(t_base + (i >> 6)) * LEAVES + (i & 63)];
    if (tid < TILE_N) {
        const int gt = t_base + tid / DEPTH_;
        const int dd = tid % DEPTH_;
        thrS[tid] = thresholds[gt * DEPTH_ + dd];
        itmp[tid] = __expf(-log_temps[gt * DEPTH_ + dd]);
    }
    __syncthreads();

    // ---- tail: one (batch row, tree) per thread ----
    const int m = tid >> 4;           // 0..15 batch row in tile
    const int j = tid & 15;           // 0..15 tree in tile
    float pos[DEPTH_], neg[DEPTH_];
    #pragma unroll
    for (int dd = 0; dd < DEPTH_; ++dd) {
        const float tl = (fvt[m * TILE_N + j * DEPTH_ + dd] - thrS[j * DEPTH_ + dd])
                         * itmp[j * DEPTH_ + dd];
        pos[dd] = fminf(fmaxf(0.5f * tl + 0.5f, 0.0f), 1.0f);   // sparsemoid( tl)
        neg[dd] = fminf(fmaxf(0.5f - 0.5f * tl, 0.0f), 1.0f);   // sparsemoid(-tl)
    }
    float accum = 0.0f;
    const float* rj = resp + j * LEAVES;
    for (int c = 0; c < LEAVES; ++c) {
        float w = 1.0f;
        #pragma unroll
        for (int dd = 0; dd < DEPTH_; ++dd)
            w *= ((c >> dd) & 1) ? neg[dd] : pos[dd];
        accum += w * rj[c];
    }
    out[(b_base + m) * NTREES + (t_base + j)] = accum;
}

// ---------------------------------------------------------------------------
extern "C" void kernel_launch(void* const* d_in, const int* in_sizes, int n_in,
                              void* d_out, int out_size, void* d_ws, size_t ws_size,
                              hipStream_t stream) {
    const float* x        = (const float*)d_in[0];  // [2048, 256]
    const float* response = (const float*)d_in[1];  // [256, 1, 64]
    const float* fsl      = (const float*)d_in[2];  // [256, 256, 6]
    const float* thr      = (const float*)d_in[3];  // [256, 6]
    const float* logt     = (const float*)d_in[4];  // [256, 6]
    float* fs  = (float*)d_ws;                      // [256, 1536] scratch
    float* out = (float*)d_out;                     // [2048, 256]

    sparsemax_kernel<<<NTOT, 256, 0, stream>>>(fsl, fs);

    const dim3 grid(BATCH_ / TILE_M, NTOT / TILE_N);
    const size_t shmem = (size_t)(TILE_M * FDIM + TILE_N * FDIM) * sizeof(__bf16); // 57344 B
    odst_fused_kernel<<<grid, 256, shmem, stream>>>(x, fs, response, thr, logt, out);
}
